// BlocksCore_46600395162113
// MI455X (gfx1250) — compile-verified, matrix operand order checked by reading, not run on hardware
//
#include <hip/hip_runtime.h>
#include <hip/hip_bf16.h>

// ---------------------------------------------------------------------------
// Types for CDNA5 WMMA (wave32): v_wmma_f32_16x16x32_bf16
// ---------------------------------------------------------------------------
typedef __bf16 bf16;
typedef __attribute__((ext_vector_type(16))) __bf16 bf16x16;
typedef __attribute__((ext_vector_type(8)))  __bf16 bf16x8;
typedef __attribute__((ext_vector_type(8)))  float  f32x8;
typedef int v4i_t __attribute__((vector_size(16)));   // matches async-LDS builtin param

// Problem constants
#define BATCH   4096
#define NHID    1024
#define NBOUT   8
#define BSOUT   128
#define NBIN    2
#define BSIN    512
#define DKIN    64
#define NINP    4096

// ---------------------------------------------------------------------------
// Optional CDNA5 async global->LDS copy (guarded; falls back to load+store)
// ---------------------------------------------------------------------------
#if defined(__has_builtin)
#  if __has_builtin(__builtin_amdgcn_global_load_async_to_lds_b128)
#    define HAVE_ASYNC_LDS 1
#  endif
#endif

__device__ __forceinline__ void stage16(void* lds_dst, const void* gsrc) {
#ifdef HAVE_ASYNC_LDS
    __builtin_amdgcn_global_load_async_to_lds_b128(
        (v4i_t*)gsrc, (v4i_t*)lds_dst, /*offset=*/0, /*cpol=*/0);
#else
    *reinterpret_cast<uint4*>(lds_dst) = *reinterpret_cast<const uint4*>(gsrc);
#endif
}

__device__ __forceinline__ void stage_wait() {
#ifdef HAVE_ASYNC_LDS
#  if __has_builtin(__builtin_amdgcn_s_wait_asynccnt)
    __builtin_amdgcn_s_wait_asynccnt(0);
#  else
    asm volatile("s_wait_asynccnt 0x0" ::: "memory");
#  endif
#endif
    __syncthreads();
}

// Load one 16-lane-striped bf16 fragment half-pair:
// lane holds K-values [base..base+7] and [base+16..base+23] (ISA 16-bit layout).
__device__ __forceinline__ bf16x16 load_frag(const bf16* base) {
    union { bf16x16 v; bf16x8 h[2]; } u;
    u.h[0] = *reinterpret_cast<const bf16x8*>(base);
    u.h[1] = *reinterpret_cast<const bf16x8*>(base + 16);
    return u.v;
}

__device__ __forceinline__ f32x8 wmma_bf16(bf16x16 a, bf16x16 b, f32x8 c) {
    return __builtin_amdgcn_wmma_f32_16x16x32_bf16(
        /*neg_a=*/false, a, /*neg_b=*/false, b,
        /*c_mod=*/(short)0, c, /*reuse_a=*/false, /*reuse_b=*/false);
}

__device__ __forceinline__ float sigm(float x) { return 1.0f / (1.0f + __expf(-x)); }

// ---------------------------------------------------------------------------
// Packing kernels: fp32 -> bf16, weights into [N][K] B-operand layouts
// ---------------------------------------------------------------------------
__global__ void k_cvt_bf16(bf16* __restrict__ dst, const float* __restrict__ src, int n) {
    int i = blockIdx.x * 256 + threadIdx.x;
    if (i < n) dst[i] = (bf16)src[i];
}

// Wv_i (512 in, 512 out) -> Wv_t[n][k] = Wv_i[k][n]
__global__ void k_pack_wv(bf16* __restrict__ dst, const float* __restrict__ wv) {
    int i = blockIdx.x * 256 + threadIdx.x;          // n*512 + k
    int n = i >> 9, k = i & 511;
    dst[i] = (bf16)wv[k * 512 + n];
}

// W_ih (4096 out, 4096 in) block-diagonal -> [j][n(512=g*128+r)][k(512)]
__global__ void k_pack_wih(bf16* __restrict__ dst, const float* __restrict__ wih) {
    int i = blockIdx.x * 256 + threadIdx.x;          // ((j*512)+n)*512 + k
    int j = i >> 18, rem = i & 262143, n = rem >> 9, k = rem & 511;
    int row = (n >> 7) * 1024 + j * 128 + (n & 127);
    dst[i] = (bf16)wih[(size_t)row * 4096 + j * 512 + k];
}

// W_hh (4096 out, 1024 in) block-diagonal -> [j][n(512)][k(128)]
__global__ void k_pack_whh(bf16* __restrict__ dst, const float* __restrict__ whh) {
    int i = blockIdx.x * 256 + threadIdx.x;          // ((j*512)+n)*128 + k
    int j = i >> 16, rem = i & 65535, n = rem >> 7, k = rem & 127;
    int row = (n >> 7) * 1024 + j * 128 + (n & 127);
    dst[i] = (bf16)whh[(size_t)row * 1024 + j * 128 + k];
}

// Wf_c / Wg_c (128 in, 128 out) -> [n(256: f then g)][k(128)]
__global__ void k_pack_wfg(bf16* __restrict__ dst, const float* __restrict__ wf,
                           const float* __restrict__ wg) {
    int i = blockIdx.x * 256 + threadIdx.x;          // n*128 + k
    int n = i >> 7, k = i & 127;
    dst[i] = (bf16)((n < 128) ? wf[k * 128 + n] : wg[k * 128 + (n - 128)]);
}

// ---------------------------------------------------------------------------
// K1: input attention + mask + (v = x@Wv_i via WMMA) fused with inp_use
// One workgroup = 16 samples, 256 threads (8 wave32).
// ---------------------------------------------------------------------------
__global__ __launch_bounds__(256) void k_input(
    const float* __restrict__ inp, const float* __restrict__ hx,
    const float* __restrict__ Wq_i, const float* __restrict__ bq_i,
    const float* __restrict__ Wk_i, const float* __restrict__ bk_i,
    const float* __restrict__ bv_i,
    const bf16* __restrict__ x_bf, const bf16* __restrict__ Wv_t,
    float* __restrict__ maskb_g, bf16* __restrict__ inpuse_g,
    float* __restrict__ out_mask, float* __restrict__ out_bm)
{
    __shared__ float kL[16][2][64];
    __shared__ float iattL[16][8][2];
    __shared__ float maskbL[16][8];

    const int t  = threadIdx.x;
    const int s0 = blockIdx.x * 16;

    // ---- phase A: k = x @ Wk_i + bk_i  (16*2*64 outputs, K=512) ----
    for (int it = 0; it < 8; ++it) {
        int o = t + it * 256;                 // 0..2047
        int m = o >> 7, i = (o >> 6) & 1, d = o & 63;
        const float* xr = inp + (size_t)(s0 + m) * NHID + i * BSIN;
        float acc = bk_i[d];
        for (int c = 0; c < BSIN; ++c) acc += xr[c] * Wk_i[c * DKIN + d];
        kL[m][i][d] = acc;
    }
    __syncthreads();

    // ---- phase B: q on-the-fly, scores, 2-way softmax ----
    if (t < 128) {
        int m = t >> 3, jb = t & 7;
        const float* hr = hx + (size_t)(s0 + m) * NHID + jb * BSOUT;
        float sA = 0.f, sB = 0.f;
        for (int d = 0; d < DKIN; ++d) {
            float q = bq_i[d];
            for (int c = 0; c < BSOUT; ++c) q += hr[c] * Wq_i[c * DKIN + d];
            sA += q * kL[m][0][d];
            sB += q * kL[m][1][d];
        }
        sA *= 0.125f; sB *= 0.125f;          // / sqrt(64)
        float mx = fmaxf(sA, sB);
        float e0 = __expf(sA - mx), e1 = __expf(sB - mx);
        float inv = 1.f / (e0 + e1);
        iattL[m][jb][0] = e0 * inv;
        iattL[m][jb][1] = e1 * inv;
    }
    __syncthreads();

    // ---- phase C: bottom-4 of iatt[:, :, 0] -> mask (lowest index wins ties) ----
    if (t < 16) {
        int m = t;
        float a[8]; bool sel[8];
        for (int j = 0; j < 8; ++j) { a[j] = iattL[m][j][0]; sel[j] = false; }
        for (int p = 0; p < 4; ++p) {
            int best = -1; float bv = 0.f;
            for (int j = 0; j < 8; ++j)
                if (!sel[j] && (best < 0 || a[j] < bv)) { best = j; bv = a[j]; }
            sel[best] = true;
        }
        for (int j = 0; j < 8; ++j) {
            float mval = sel[j] ? 0.f : 1.f;
            maskbL[m][j] = mval;
            maskb_g[(size_t)(s0 + m) * 8 + j] = mval;
            out_bm [(size_t)(s0 + m) * 8 + j] = mval;   // block_mask output
        }
    }
    __syncthreads();

    // mask output (B, 1024): repeat per block
    for (int it = 0; it < 64; ++it) {
        int idx = t + it * 256;               // 0..16383
        int m = idx >> 10, col = idx & 1023;
        out_mask[(size_t)(s0 + m) * NHID + col] = maskbL[m][col >> 7];
    }

    // ---- phase D: WMMA  v = x@Wv_i,  inp_use = iatt0*v0 + iatt1*v1 ----
    const int lane  = t & 31, w = t >> 5;
    const int nlan  = lane & 15;
    const int koff8 = (lane >> 4) * 8;
    for (int nt4 = 0; nt4 < 4; ++nt4) {
        int n0 = (w * 4 + nt4) * 16;
        f32x8 c0 = {}; f32x8 c1 = {};
        const bf16* arow0 = x_bf + (size_t)(s0 + nlan) * NHID;       // block i=0
        const bf16* arow1 = arow0 + BSIN;                            // block i=1
        const bf16* brow  = Wv_t + (size_t)(n0 + nlan) * 512;
        __builtin_prefetch(brow, 0, 3);
        #pragma unroll 4
        for (int kt = 0; kt < 16; ++kt) {
            int kb = kt * 32 + koff8;
            bf16x16 bf = load_frag(brow  + kb);
            bf16x16 a0 = load_frag(arow0 + kb);
            bf16x16 a1 = load_frag(arow1 + kb);
            c0 = wmma_bf16(a0, bf, c0);
            c1 = wmma_bf16(a1, bf, c1);
        }
        int n = n0 + nlan;
        float bv = bv_i[n];
        for (int vv = 0; vv < 8; ++vv) {
            int m = ((lane >> 4) << 3) + vv;
            float v0 = c0[vv] + bv;
            float v1 = c1[vv] + bv;
            bf16* orow = inpuse_g + (size_t)(s0 + m) * NINP;
            for (int jb = 0; jb < 8; ++jb) {
                float val = iattL[m][jb][0] * v0 + iattL[m][jb][1] * v1;
                orow[jb * 512 + n] = (bf16)val;
            }
        }
    }
}

// ---------------------------------------------------------------------------
// K2: gates (block-sparse WMMA, K=512 ih + K=128 hh) + LSTM + cx_out
// grid = (B/32, 8 blocks), 256 threads. 32 samples per WG = two M-tiles so
// each weight (B) fragment feeds two WMMAs; A-tiles staged once in LDS
// (async global->LDS when available) and re-read via ds_load_b128.
// ---------------------------------------------------------------------------
__global__ __launch_bounds__(256) void k_gates(
    const bf16* __restrict__ inpuse, const bf16* __restrict__ hx_bf,
    const bf16* __restrict__ wih_t,  const bf16* __restrict__ whh_t,
    const float* __restrict__ b_ih,  const float* __restrict__ b_hh,
    const float* __restrict__ cx,    const float* __restrict__ maskb,
    float* __restrict__ cxnew, float* __restrict__ hb_f32,
    bf16* __restrict__ hb_bf,  float* __restrict__ out_cx)
{
    __shared__ __align__(16) char smem[65536];
    bf16*  aih  = (bf16*)smem;             // [32][512] bf16 = 32 KB (stage phase)
    bf16*  ahh  = (bf16*)(smem + 32768);   // [32][128] bf16 =  8 KB (stage phase)
    float* gacc = (float*)smem;            // [32][512] f32  = 64 KB (after barrier)

    const int t = threadIdx.x, lane = t & 31, w = t >> 5;
    const int s0 = blockIdx.x * 32, j = blockIdx.y;
    const int nlan = lane & 15, koff8 = (lane >> 4) * 8;

    // ---- stage A-tiles into LDS: inp_use 32x512 bf16, hx 32x128 bf16 ----
    for (int it = 0; it < 8; ++it) {
        int e = t + it * 256;             // 0..2047 16B-chunks
        int m = e >> 6, q = e & 63;
        stage16(aih + m * 512 + q * 8,
                inpuse + (size_t)(s0 + m) * NINP + j * 512 + q * 8);
    }
    for (int it = 0; it < 2; ++it) {
        int e = t + it * 256;             // 0..511 16B-chunks
        int m = e >> 4, q = e & 15;
        stage16(ahh + m * 128 + q * 8,
                hx_bf + (size_t)(s0 + m) * NHID + j * BSOUT + q * 8);
    }
    stage_wait();

    // ---- WMMA: accumulate 4 N-tiles x 2 M-tiles in registers ----
    f32x8 acc[4][2];
    #pragma unroll
    for (int nt4 = 0; nt4 < 4; ++nt4) { acc[nt4][0] = (f32x8){}; acc[nt4][1] = (f32x8){}; }

    for (int nt4 = 0; nt4 < 4; ++nt4) {
        int n0 = (w * 4 + nt4) * 16;
        const bf16* brow  = wih_t + ((size_t)(j * 512 + n0 + nlan)) * 512;
        const bf16* brow2 = whh_t + ((size_t)(j * 512 + n0 + nlan)) * 128;
        __builtin_prefetch(brow, 0, 3);
        const bf16* a0 = aih + (size_t)nlan * 512;
        const bf16* a1 = aih + (size_t)(16 + nlan) * 512;
        #pragma unroll 4
        for (int kt = 0; kt < 16; ++kt) {
            int kb = kt * 32 + koff8;
            bf16x16 bf = load_frag(brow + kb);
            acc[nt4][0] = wmma_bf16(load_frag(a0 + kb), bf, acc[nt4][0]);
            acc[nt4][1] = wmma_bf16(load_frag(a1 + kb), bf, acc[nt4][1]);
        }
        const bf16* h0 = ahh + (size_t)nlan * 128;
        const bf16* h1 = ahh + (size_t)(16 + nlan) * 128;
        #pragma unroll
        for (int kt = 0; kt < 4; ++kt) {
            int kb = kt * 32 + koff8;
            bf16x16 bf = load_frag(brow2 + kb);
            acc[nt4][0] = wmma_bf16(load_frag(h0 + kb), bf, acc[nt4][0]);
            acc[nt4][1] = wmma_bf16(load_frag(h1 + kb), bf, acc[nt4][1]);
        }
    }
    __syncthreads();   // all waves done reading A LDS; reuse region for gacc

    for (int nt4 = 0; nt4 < 4; ++nt4) {
        int n0 = (w * 4 + nt4) * 16;
        for (int vv = 0; vv < 8; ++vv) {
            int mrow = ((lane >> 4) << 3) + vv;
            gacc[(size_t)mrow * 512 + n0 + nlan]        = acc[nt4][0][vv];
            gacc[(size_t)(16 + mrow) * 512 + n0 + nlan] = acc[nt4][1][vv];
        }
    }
    __syncthreads();

    // ---- LSTM elementwise; gate order i,f,g,o at n = g*128 + r ----
    for (int it = 0; it < 16; ++it) {
        int idx = t + it * 256;               // 0..4095
        int m = idx >> 7, r = idx & 127;
        int gidx = j * BSOUT + r;
        float gi = gacc[m * 512 +       r] + b_ih[       gidx] + b_hh[       gidx];
        float gf = gacc[m * 512 + 128 + r] + b_ih[1024 + gidx] + b_hh[1024 + gidx];
        float gg = gacc[m * 512 + 256 + r] + b_ih[2048 + gidx] + b_hh[2048 + gidx];
        float go = gacc[m * 512 + 384 + r] + b_ih[3072 + gidx] + b_hh[3072 + gidx];
        size_t off = (size_t)(s0 + m) * NHID + gidx;
        float cxv = cx[off];
        float cn  = sigm(gf) * cxv + sigm(gi) * tanhf(gg);
        float hl  = sigm(go) * tanhf(cn);
        cxnew[off]  = cn;
        hb_f32[off] = hl;
        hb_bf[off]  = (bf16)hl;
        float mk = maskb[(size_t)(s0 + m) * 8 + j];
        out_cx[off] = mk * cn + (1.f - mk) * cxv;
    }
}

// ---------------------------------------------------------------------------
// K3: communication attention (q/k/v small GEMMs + 4-head softmax, VALU),
// out -> cout_bf. One workgroup = 16 samples.
// ---------------------------------------------------------------------------
__global__ __launch_bounds__(256) void k_comm(
    const bf16* __restrict__ hb_bf,
    const float* __restrict__ Wq_c, const float* __restrict__ bq_c,
    const float* __restrict__ Wk_c, const float* __restrict__ bk_c,
    const float* __restrict__ Wv_c, const float* __restrict__ bv_c,
    bf16* __restrict__ cout_bf)
{
    __shared__ bf16 kvL[16][8][128];
    const int t = threadIdx.x;
    const int s0 = blockIdx.x * 16;
    const float invs = 0.17677669529663687f;   // 1/sqrt(32)

    // phase 1: k = hb @ Wk_c + bk_c  -> kvL
    for (int it = 0; it < 64; ++it) {
        int o = t + it * 256;                  // 0..16383
        int m = o >> 10, kb = (o >> 7) & 7, d = o & 127;
        const bf16* hr = hb_bf + (size_t)(s0 + m) * NHID + kb * BSOUT;
        float acc = bk_c[d];
        for (int c = 0; c < 128; ++c) acc += (float)hr[c] * Wk_c[c * 128 + d];
        kvL[m][kb][d] = (bf16)acc;
    }
    __syncthreads();

    // phase 2: q on the fly, scores over 8 blocks, softmax
    float attn[2][8];
    for (int half = 0; half < 2; ++half) {
        int it = t + half * 256;               // 0..511
        int m = it >> 5, h = (it >> 3) & 3, qb = it & 7;
        const bf16* hr = hb_bf + (size_t)(s0 + m) * NHID + qb * BSOUT;
        float qv[32];
        for (int dd = 0; dd < 32; ++dd) {
            int n = h * 32 + dd;
            float acc = bq_c[n];
            for (int c = 0; c < 128; ++c) acc += (float)hr[c] * Wq_c[c * 128 + n];
            qv[dd] = acc;
        }
        float sc[8], mx = -1e30f;
        for (int kb = 0; kb < 8; ++kb) {
            float s = 0.f;
            for (int dd = 0; dd < 32; ++dd) s += qv[dd] * (float)kvL[m][kb][h * 32 + dd];
            s *= invs;
            sc[kb] = s; mx = fmaxf(mx, s);
        }
        float den = 0.f;
        for (int kb = 0; kb < 8; ++kb) { sc[kb] = __expf(sc[kb] - mx); den += sc[kb]; }
        float inv = 1.f / den;
        for (int kb = 0; kb < 8; ++kb) attn[half][kb] = sc[kb] * inv;
    }
    __syncthreads();

    // phase 3: v = hb @ Wv_c + bv_c  (overwrite kvL)
    for (int it = 0; it < 64; ++it) {
        int o = t + it * 256;
        int m = o >> 10, kb = (o >> 7) & 7, d = o & 127;
        const bf16* hr = hb_bf + (size_t)(s0 + m) * NHID + kb * BSOUT;
        float acc = bv_c[d];
        for (int c = 0; c < 128; ++c) acc += (float)hr[c] * Wv_c[c * 128 + d];
        kvL[m][kb][d] = (bf16)acc;
    }
    __syncthreads();

    // phase 4: out[m][qb][h*32+dd] = sum_kb attn*v
    for (int half = 0; half < 2; ++half) {
        int it = t + half * 256;
        int m = it >> 5, h = (it >> 3) & 3, qb = it & 7;
        for (int dd = 0; dd < 32; ++dd) {
            float o = 0.f;
            for (int kb = 0; kb < 8; ++kb)
                o += attn[half][kb] * (float)kvL[m][kb][h * 32 + dd];
            cout_bf[(size_t)(s0 + m) * NHID + qb * BSOUT + h * 32 + dd] = (bf16)o;
        }
    }
}

// ---------------------------------------------------------------------------
// K4: comm gating (out@Wf_c, out@Wg_c via WMMA) + hx_new + final hx masking.
// ---------------------------------------------------------------------------
__global__ __launch_bounds__(256) void k_final(
    const bf16* __restrict__ cout_bf, const bf16* __restrict__ wfg_t,
    const float* __restrict__ bf_c, const float* __restrict__ bg_c,
    const float* __restrict__ hb_f32, const float* __restrict__ maskb,
    const float* __restrict__ hx, float* __restrict__ out_hx)
{
    __shared__ float fgL[16][256];
    const int t = threadIdx.x, lane = t & 31, w = t >> 5;
    const int s0 = blockIdx.x * 16;
    const int nlan = lane & 15, koff8 = (lane >> 4) * 8;

    for (int jb = 0; jb < 8; ++jb) {
        for (int nt2 = 0; nt2 < 2; ++nt2) {
            int n0 = (w * 2 + nt2) * 16;
            f32x8 c = {};
            const bf16* arow = cout_bf + (size_t)(s0 + nlan) * NHID + jb * BSOUT;
            const bf16* brow = wfg_t + (size_t)(n0 + nlan) * 128;
            #pragma unroll
            for (int kt = 0; kt < 4; ++kt) {
                int kb = kt * 32 + koff8;
                c = wmma_bf16(load_frag(arow + kb), load_frag(brow + kb), c);
            }
            for (int vv = 0; vv < 8; ++vv)
                fgL[((lane >> 4) << 3) + vv][n0 + nlan] = c[vv];
        }
        __syncthreads();
        for (int it = 0; it < 8; ++it) {
            int idx = t + it * 256;           // 0..2047
            int m = idx >> 7, r = idx & 127;
            float fv = fgL[m][r]       + bf_c[r];
            float gv = fgL[m][128 + r] + bg_c[r];
            float comm = sigm(gv) * tanhf(fv);
            size_t off = (size_t)(s0 + m) * NHID + jb * BSOUT + r;
            float hxn = hb_f32[off] + comm;
            float mk  = maskb[(size_t)(s0 + m) * 8 + jb];
            out_hx[off] = mk * hxn + (1.f - mk) * hx[off];
        }
        __syncthreads();
    }
}

// ---------------------------------------------------------------------------
// Host launcher
// ---------------------------------------------------------------------------
extern "C" void kernel_launch(void* const* d_in, const int* in_sizes, int n_in,
                              void* d_out, int out_size, void* d_ws, size_t ws_size,
                              hipStream_t stream) {
    const float* inp  = (const float*)d_in[0];
    const float* hx   = (const float*)d_in[1];
    const float* cx   = (const float*)d_in[2];
    const float* Wq_i = (const float*)d_in[4];
    const float* bq_i = (const float*)d_in[5];
    const float* Wk_i = (const float*)d_in[6];
    const float* bk_i = (const float*)d_in[7];
    const float* Wv_i = (const float*)d_in[8];
    const float* bv_i = (const float*)d_in[9];
    const float* Wq_c = (const float*)d_in[10];
    const float* bq_c = (const float*)d_in[11];
    const float* Wk_c = (const float*)d_in[12];
    const float* bk_c = (const float*)d_in[13];
    const float* Wv_c = (const float*)d_in[14];
    const float* bv_c = (const float*)d_in[15];
    const float* Wf_c = (const float*)d_in[16];
    const float* bf_c = (const float*)d_in[17];
    const float* Wg_c = (const float*)d_in[18];
    const float* bg_c = (const float*)d_in[19];
    const float* W_ih = (const float*)d_in[20];
    const float* b_ih = (const float*)d_in[21];
    const float* W_hh = (const float*)d_in[22];
    const float* b_hh = (const float*)d_in[23];

    float* out  = (float*)d_out;
    float* o_hx = out;
    float* o_cx = out + (size_t)BATCH * NHID;
    float* o_mk = out + (size_t)BATCH * NHID * 2;
    float* o_bm = out + (size_t)BATCH * NHID * 3;

    char* ws = (char*)d_ws;
    bf16* wv_t    = (bf16*)(ws + 0x00000000);   // 512 KB
    bf16* wih_t   = (bf16*)(ws + 0x00080000);   // 4 MB
    bf16* whh_t   = (bf16*)(ws + 0x00480000);   // 1 MB
    bf16* wfg_t   = (bf16*)(ws + 0x00580000);   // 64 KB
    bf16* x_bf    = (bf16*)(ws + 0x005A0000);   // 8 MB
    bf16* hx_bf   = (bf16*)(ws + 0x00DA0000);   // 8 MB
    float* maskb  = (float*)(ws + 0x015E0000);  // 128 KB
    bf16* inpuse  = (bf16*)(ws + 0x01600000);   // 32 MB
    float* cxnew  = (float*)(ws + 0x03600000);  // 16 MB
    float* hb_f32 = (float*)(ws + 0x04600000);  // 16 MB
    bf16* hb_bf   = (bf16*)(ws + 0x05600000);   // 8 MB
    bf16* cout_bf = (bf16*)(ws + 0x05E00000);   // 8 MB

    // --- pack / convert ---
    k_cvt_bf16<<<(BATCH * NHID) / 256, 256, 0, stream>>>(x_bf, inp, BATCH * NHID);
    k_cvt_bf16<<<(BATCH * NHID) / 256, 256, 0, stream>>>(hx_bf, hx, BATCH * NHID);
    k_pack_wv <<<(512 * 512) / 256, 256, 0, stream>>>(wv_t, Wv_i);
    k_pack_wih<<<(8 * 512 * 512) / 256, 256, 0, stream>>>(wih_t, W_ih);
    k_pack_whh<<<(8 * 512 * 128) / 256, 256, 0, stream>>>(whh_t, W_hh);
    k_pack_wfg<<<(256 * 128) / 256, 256, 0, stream>>>(wfg_t, Wf_c, Wg_c);

    // --- pipeline ---
    k_input<<<BATCH / 16, 256, 0, stream>>>(inp, hx, Wq_i, bq_i, Wk_i, bk_i, bv_i,
                                            x_bf, wv_t, maskb, inpuse, o_mk, o_bm);
    k_gates<<<dim3(BATCH / 32, 8), 256, 0, stream>>>(inpuse, hx_bf, wih_t, whh_t,
                                                     b_ih, b_hh, cx, maskb,
                                                     cxnew, hb_f32, hb_bf, o_cx);
    k_comm<<<BATCH / 16, 256, 0, stream>>>(hb_bf, Wq_c, bq_c, Wk_c, bk_c, Wv_c, bv_c,
                                           cout_bf);
    k_final<<<BATCH / 16, 256, 0, stream>>>(cout_bf, wfg_t, bf_c, bg_c,
                                            hb_f32, maskb, hx, o_hx);
}
